// NetworkAwareAttention_23115513987696
// MI455X (gfx1250) — matrix operand order repaired and checked by reading, not compiled, not run on hardware
//
#include <hip/hip_runtime.h>

// ---------------- problem constants ----------------
#define BATCH 2
#define SEQ   2048
#define DIM   1024
#define NHEAD 16
#define HDIM  64
#define BS    (BATCH * SEQ)          // 4096 tokens
static constexpr float SM_SCALE  = 1.35f / 8.0f;   // focus_factor / sqrt(hd)
static constexpr float NEG_BIG   = -1.35e9f;       // (-1e9) * focus_factor
static constexpr float INV_HEADS = 1.0f / (float)NHEAD;

// ---------------- vector types ----------------
typedef __attribute__((ext_vector_type(16))) __bf16 v16bf;
typedef __attribute__((ext_vector_type(8)))  __bf16 v8bf;
typedef __attribute__((ext_vector_type(8)))  float  v8f;

#define WMMA_BF16(A, B, C) \
  __builtin_amdgcn_wmma_f32_16x16x32_bf16(false, (A), false, (B), (short)0, (C), false, false)

// A-fragment (16x32 bf16, M x K): lane L holds row M=L%16,
// K = (L/16)*16 + [0..15] in elems 0..15 (contiguous).
__device__ inline v16bf load_a_frag(const __bf16* __restrict__ rowp) {
  v8bf lo = *(const v8bf*)rowp;
  v8bf hi = *(const v8bf*)(rowp + 8);
  v16bf a;
#pragma unroll
  for (int i = 0; i < 8; ++i) { a[i] = lo[i]; a[8 + i] = hi[i]; }
  return a;
}

// B-fragment (32x16 bf16, K x N): lane L holds col N=L%16,
// K = (L/16)*16 + [0..15] in elems 0..15 (contiguous).
__device__ inline v16bf load_b_frag(const __bf16* __restrict__ colp) {
  v8bf lo = *(const v8bf*)colp;
  v8bf hi = *(const v8bf*)(colp + 8);
  v16bf b;
#pragma unroll
  for (int i = 0; i < 8; ++i) { b[i] = lo[i]; b[8 + i] = hi[i]; }
  return b;
}

// ---------------- helper kernels ----------------
__global__ __launch_bounds__(256) void cvt_f32_bf16(const float* __restrict__ in,
                                                    __bf16* __restrict__ out, int n) {
  int i = blockIdx.x * 256 + threadIdx.x;
  if (i < n) out[i] = (__bf16)in[i];
}

__global__ __launch_bounds__(256) void zero_f32(float* __restrict__ p, int n) {
  int i = blockIdx.x * 256 + threadIdx.x;
  if (i < n) p[i] = 0.0f;
}

// ---------------- Q/K projection ----------------
// dst[channel m][token n] tile, stored to dst[B,H,S,HD] (bf16).
// A = W (M=out-channel, K=in-channel), B = x^T (K=in-channel, N=token).
__global__ __launch_bounds__(256) void proj_qk_kernel(const __bf16* __restrict__ xb,
                                                      const __bf16* __restrict__ Wb,
                                                      const float*  __restrict__ bias,
                                                      __bf16* __restrict__ dst) {
  const int lane = threadIdx.x & 31;
  const int wid  = threadIdx.x >> 5;
  const int lr   = lane & 15;     // low-lane index
  const int lh   = lane >> 4;     // lane half (0/1)
  const int n0 = (blockIdx.x * 8 + wid) * 16;   // token tile base   (0..4095)
  const int m0 = blockIdx.y * 16;               // channel tile base (0..1023)

  v8f acc = {};
  for (int k0 = 0; k0 < DIM; k0 += 32) {
    v16bf a = load_a_frag(Wb + (size_t)(m0 + lr) * DIM + k0 + lh * 16);
    v16bf b = load_b_frag(xb + (size_t)(n0 + lr) * DIM + k0 + lh * 16);
    acc = WMMA_BF16(a, b, acc);
  }
  const int tok = n0 + lr;                       // this lane's token
  const int bb  = tok >> 11;                     // /SEQ
  const int ss  = tok & (SEQ - 1);
  const int ch0 = m0 + lh * 8;                   // 8 contiguous channels
  const int h   = ch0 >> 6;
  const int d0  = ch0 & (HDIM - 1);
  v8bf o;
#pragma unroll
  for (int v = 0; v < 8; ++v) o[v] = (__bf16)(acc[v] + bias[ch0 + v]);
  *(v8bf*)(dst + (((size_t)(bb * NHEAD + h) * SEQ + ss) * HDIM + d0)) = o;
}

// ---------------- V projection (stores V^T) ----------------
// A = x (M=token), B = W^T (N=channel). D tile [token][channel] -> Vt[B,H,HD,S].
__global__ __launch_bounds__(256) void proj_v_kernel(const __bf16* __restrict__ xb,
                                                     const __bf16* __restrict__ Wb,
                                                     const float*  __restrict__ bias,
                                                     __bf16* __restrict__ vt) {
  const int lane = threadIdx.x & 31;
  const int wid  = threadIdx.x >> 5;
  const int lr   = lane & 15;
  const int lh   = lane >> 4;
  const int m0 = (blockIdx.x * 8 + wid) * 16;   // token tile
  const int n0 = blockIdx.y * 16;               // channel tile

  v8f acc = {};
  for (int k0 = 0; k0 < DIM; k0 += 32) {
    v16bf a = load_a_frag(xb + (size_t)(m0 + lr) * DIM + k0 + lh * 16);
    v16bf b = load_b_frag(Wb + (size_t)(n0 + lr) * DIM + k0 + lh * 16);
    acc = WMMA_BF16(a, b, acc);
  }
  const int ch = n0 + lr;                        // this lane's channel
  const int h  = ch >> 6;
  const int d  = ch & (HDIM - 1);
  const float bv = bias[ch];
  const int t0 = m0 + lh * 8;                    // 8 contiguous tokens
  const int bb = t0 >> 11;
  const int s0 = t0 & (SEQ - 1);
  v8bf o;
#pragma unroll
  for (int v = 0; v < 8; ++v) o[v] = (__bf16)(acc[v] + bv);
  *(v8bf*)(vt + (((size_t)(bb * NHEAD + h) * HDIM + d) * SEQ + s0)) = o;
}

// ---------------- attention core ----------------
// One block = (b, h, 32 query rows). Scores 32 x 2048 f32 live in LDS (256 KB).
__global__ __launch_bounds__(256) void attn_kernel(const __bf16* __restrict__ Qb,
                                                   const __bf16* __restrict__ Kb,
                                                   const __bf16* __restrict__ Vtb,
                                                   const int*    __restrict__ mask,
                                                   __bf16* __restrict__ attb,
                                                   float*  __restrict__ avg) {
  extern __shared__ float sc[];                  // [32][SEQ]
  const int lane = threadIdx.x & 31;
  const int wid  = threadIdx.x >> 5;
  const int lr   = lane & 15;
  const int lh   = lane >> 4;
  const int qt = blockIdx.x & 63;
  const int h  = (blockIdx.x >> 6) & 15;
  const int b  = blockIdx.x >> 10;
  const int q0 = qt * 32;
  const int bh = b * NHEAD + h;

  // ---- Q fragments for both 16-row subtiles, both K-steps over hd=64 ----
  v16bf qa[2][2];
#pragma unroll
  for (int mt = 0; mt < 2; ++mt)
#pragma unroll
    for (int ks = 0; ks < 2; ++ks)
      qa[mt][ks] = load_a_frag(Qb + ((size_t)(bh * SEQ + q0 + mt * 16 + lr)) * HDIM
                                  + ks * 32 + lh * 16);

  // ---- scores = (Q K^T) * SM_SCALE, masked, into LDS ----
  for (int n0 = wid * 16; n0 < SEQ; n0 += 128) {
    v16bf kf[2];
#pragma unroll
    for (int ks = 0; ks < 2; ++ks)
      kf[ks] = load_b_frag(Kb + ((size_t)(bh * SEQ + n0 + lr)) * HDIM + ks * 32 + lh * 16);
    const int col = n0 + lr;
#pragma unroll
    for (int mt = 0; mt < 2; ++mt) {
      v8f acc = {};
      acc = WMMA_BF16(qa[mt][0], kf[0], acc);
      acc = WMMA_BF16(qa[mt][1], kf[1], acc);
#pragma unroll
      for (int v = 0; v < 8; ++v) {
        const int r = mt * 16 + lh * 8 + v;      // local q row 0..31
        const int mv = mask[(size_t)b * SEQ * SEQ + (size_t)(q0 + r) * SEQ + col];
        sc[r * SEQ + col] = mv ? acc[v] * SM_SCALE : NEG_BIG;
      }
    }
  }
  __syncthreads();

  // ---- row softmax (8 lanes per row) + accumulate avg attention ----
  {
    const int r   = threadIdx.x >> 3;
    const int sub = threadIdx.x & 7;
    float* row = sc + r * SEQ;
    float mx = -3.4e38f;
    for (int c = sub; c < SEQ; c += 8) mx = fmaxf(mx, row[c]);
#pragma unroll
    for (int o = 4; o >= 1; o >>= 1) mx = fmaxf(mx, __shfl_xor(mx, o, 8));
    float sum = 0.f;
    for (int c = sub; c < SEQ; c += 8) {
      float p = __expf(row[c] - mx);
      row[c] = p; sum += p;
    }
#pragma unroll
    for (int o = 4; o >= 1; o >>= 1) sum += __shfl_xor(sum, o, 8);
    const float inv = 1.0f / sum;
    float* avgrow = avg + (size_t)b * SEQ * SEQ + (size_t)(q0 + r) * SEQ;
    for (int c = sub; c < SEQ; c += 8) {
      float p = row[c] * inv;
      row[c] = p;
      atomicAdd(&avgrow[c], p * INV_HEADS);
    }
  }
  __syncthreads();

  // ---- attended^T = V^T x P^T : wave tile = 16 dims x 16 tokens ----
  {
    const int dt = wid & 3;                      // dim tile (hd/16)
    const int tt = wid >> 2;                     // token tile (32/16)
    v8f acc = {};
    for (int k0 = 0; k0 < SEQ; k0 += 32) {
      v16bf a = load_a_frag(Vtb + ((size_t)(bh * HDIM + dt * 16 + lr)) * SEQ + k0 + lh * 16);
      // B = P^T : lane col = local token, K contiguous along score row
      const float* p = sc + (tt * 16 + lr) * SEQ + k0 + lh * 16;
      v16bf bf;
#pragma unroll
      for (int j = 0; j < 16; j += 4) {
        float4 f = *(const float4*)(p + j);
        bf[j] = (__bf16)f.x; bf[j + 1] = (__bf16)f.y;
        bf[j + 2] = (__bf16)f.z; bf[j + 3] = (__bf16)f.w;
      }
      acc = WMMA_BF16(a, bf, acc);
    }
    const int s  = q0 + tt * 16 + lr;            // this lane's token (within batch)
    const int d0 = dt * 16 + lh * 8;             // 8 contiguous head-dims
    v8bf o;
#pragma unroll
    for (int v = 0; v < 8; ++v) o[v] = (__bf16)acc[v];
    *(v8bf*)(attb + ((size_t)(b * SEQ + s)) * DIM + h * HDIM + d0) = o;
  }
}

// ---------------- output projection ----------------
// A = Wo (M=out-channel), B = att^T (N=token). f32 output + bias.
__global__ __launch_bounds__(256) void out_proj_kernel(const __bf16* __restrict__ attb,
                                                       const __bf16* __restrict__ Wob,
                                                       const float*  __restrict__ bias,
                                                       float* __restrict__ out) {
  const int lane = threadIdx.x & 31;
  const int wid  = threadIdx.x >> 5;
  const int lr   = lane & 15;
  const int lh   = lane >> 4;
  const int n0 = (blockIdx.x * 8 + wid) * 16;   // token tile
  const int m0 = blockIdx.y * 16;               // channel tile

  v8f acc = {};
  for (int k0 = 0; k0 < DIM; k0 += 32) {
    v16bf a = load_a_frag(Wob  + (size_t)(m0 + lr) * DIM + k0 + lh * 16);
    v16bf b = load_b_frag(attb + (size_t)(n0 + lr) * DIM + k0 + lh * 16);
    acc = WMMA_BF16(a, b, acc);
  }
  const int tok = n0 + lr;
  const int ch0 = m0 + lh * 8;
  float* dst = out + (size_t)tok * DIM + ch0;
#pragma unroll
  for (int v = 0; v < 8; ++v) dst[v] = acc[v] + bias[ch0 + v];
}

// ---------------- host launch ----------------
extern "C" void kernel_launch(void* const* d_in, const int* in_sizes, int n_in,
                              void* d_out, int out_size, void* d_ws, size_t ws_size,
                              hipStream_t stream) {
  const float* x    = (const float*)d_in[0];
  const int*   mask = (const int*)  d_in[1];
  const float* Wq   = (const float*)d_in[2];
  const float* bq   = (const float*)d_in[3];
  const float* Wk   = (const float*)d_in[4];
  const float* bk   = (const float*)d_in[5];
  const float* Wv   = (const float*)d_in[6];
  const float* bv   = (const float*)d_in[7];
  const float* Wo   = (const float*)d_in[8];
  const float* bo   = (const float*)d_in[9];

  float* out = (float*)d_out;                        // [B,S,D]
  float* avg = out + (size_t)BATCH * SEQ * DIM;      // [B,S,S]

  // workspace layout (bf16 buffers)
  char* w = (char*)d_ws;
  __bf16* xb  = (__bf16*)(w + 0);                    //  8 MB  x        [BS, D]
  __bf16* Wqb = (__bf16*)(w + (8u  << 20));          //  2 MB
  __bf16* Wkb = (__bf16*)(w + (10u << 20));          //  2 MB
  __bf16* Wvb = (__bf16*)(w + (12u << 20));          //  2 MB
  __bf16* Wob = (__bf16*)(w + (14u << 20));          //  2 MB
  __bf16* Qb  = (__bf16*)(w + (16u << 20));          //  8 MB  [B,H,S,hd]
  __bf16* Kb  = (__bf16*)(w + (24u << 20));          //  8 MB  [B,H,S,hd]
  __bf16* Vtb = (__bf16*)(w + (32u << 20));          //  8 MB  [B,H,hd,S]
  __bf16* atb = (__bf16*)(w + (40u << 20));          //  8 MB  [B,S,D]

  const int NX = BS * DIM;       // 4,194,304
  const int NW = DIM * DIM;      // 1,048,576
  const int NA = BATCH * SEQ * SEQ; // 8,388,608

  cvt_f32_bf16<<<(NX + 255) / 256, 256, 0, stream>>>(x,  xb,  NX);
  cvt_f32_bf16<<<(NW + 255) / 256, 256, 0, stream>>>(Wq, Wqb, NW);
  cvt_f32_bf16<<<(NW + 255) / 256, 256, 0, stream>>>(Wk, Wkb, NW);
  cvt_f32_bf16<<<(NW + 255) / 256, 256, 0, stream>>>(Wv, Wvb, NW);
  cvt_f32_bf16<<<(NW + 255) / 256, 256, 0, stream>>>(Wo, Wob, NW);
  zero_f32<<<(NA + 255) / 256, 256, 0, stream>>>(avg, NA);

  dim3 pg(BS / 16 / 8, DIM / 16);                    // (32, 64)
  proj_qk_kernel<<<pg, 256, 0, stream>>>(xb, Wqb, bq, Qb);
  proj_qk_kernel<<<pg, 256, 0, stream>>>(xb, Wkb, bk, Kb);
  proj_v_kernel <<<pg, 256, 0, stream>>>(xb, Wvb, bv, Vtb);

  // (b, h, 32-row q-tile) blocks; 256 KB dynamic LDS (fits CDNA5 320 KB WGP LDS)
  attn_kernel<<<BATCH * NHEAD * (SEQ / 32), 256, 32 * SEQ * sizeof(float), stream>>>(
      Qb, Kb, Vtb, mask, atb, avg);

  out_proj_kernel<<<pg, 256, 0, stream>>>(atb, Wob, bo, out);
}
